// RecursiveNN_31894427140538
// MI455X (gfx1250) — compile-verified, matrix operand order checked by reading, not run on hardware
//
#include <hip/hip_runtime.h>
#include <hip/hip_bf16.h>

// ---- problem constants (from reference) ----
#define BB 512   // batch
#define LL 512   // leaves
#define EE 128   // emb size
#define OPN 5    // op size
#define II 511   // internal nodes

typedef __attribute__((ext_vector_type(16))) __bf16 v16bf;
typedef __attribute__((ext_vector_type(8)))  __bf16 v8bf;
typedef __attribute__((ext_vector_type(8)))  float  v8f;

#define WMMA_BF16(a, b, c) \
  __builtin_amdgcn_wmma_f32_16x16x32_bf16(false, (a), false, (b), (short)0, (c), false, false)

// Build a 16x32 bf16 A-operand tile from an f32 row in global memory.
// Per ISA 16-bit A layout: lanes 0-15 hold K = [k0..k0+7, k0+16..k0+23] (khalf=0),
// lanes 16-31 hold K offset by +8 (khalf=8). Caller passes k0 = 32*t + khalf.
__device__ __forceinline__ v16bf packA(const float* row, int k0) {
  v16bf r;
#pragma unroll
  for (int i = 0; i < 8; ++i) {
    r[i]     = (__bf16)row[k0 + i];
    r[i + 8] = (__bf16)row[k0 + 16 + i];
  }
  return r;
}

// Same A-operand tile, from a bf16 [16 rows][128 cols] LDS buffer (two b128 reads).
__device__ __forceinline__ v16bf ldsA(const __bf16* buf, int row, int k0) {
  union { v16bf v; v8bf h[2]; } u;
  u.h[0] = *(const v8bf*)(buf + row * EE + k0);
  u.h[1] = *(const v8bf*)(buf + row * EE + k0 + 16);
  return u.v;
}

// ---------------------------------------------------------------------------
// Kernel 0: pack W_top / W_bot / padded G into per-lane v16bf B-operand layout.
// Packed index: (((w*4 + t)*32 + lane)*16 + i); element (lane,i) -> (k,n):
//   n = 16*w + (lane&15),  k = 32*t + (i&7) + (i>>3)*16 + (lane>>4)*8
// ---------------------------------------------------------------------------
__global__ void rnn_prep_kernel(const float* __restrict__ W_w,
                                const float* __restrict__ G_w,
                                __bf16* __restrict__ wtop,
                                __bf16* __restrict__ wbot,
                                __bf16* __restrict__ gpad) {
  int idx = blockIdx.x * 256 + threadIdx.x;
  if (idx < 8 * 4 * 32 * 16) {
    int i = idx & 15, lane = (idx >> 4) & 31, t = (idx >> 9) & 3, w = idx >> 11;
    int n = 16 * w + (lane & 15);
    int k = 32 * t + (i & 7) + ((i >> 3) << 4) + ((lane >> 4) << 3);
    wtop[idx] = (__bf16)W_w[(size_t)k * EE + n];          // rows 0..127 of W_w
    wbot[idx] = (__bf16)W_w[(size_t)(k + EE) * EE + n];   // rows 128..255
  }
  if (idx < 4 * 32 * 16) {  // single N-tile, cols 5..15 zero
    int i = idx & 15, lane = (idx >> 4) & 31, t = (idx >> 9) & 3;
    int n = lane & 15;
    int k = 32 * t + (i & 7) + ((i >> 3) << 4) + ((lane >> 4) << 3);
    gpad[idx] = (n < OPN) ? (__bf16)G_w[(size_t)k * OPN + n] : (__bf16)0.0f;
  }
}

// ---------------------------------------------------------------------------
// Kernel 1: preR[r][b][e] = emb[b,r,:] @ W_bot + W_b  (parallel over all b,r)
// grid = L * (B/16); 8 waves, wave w owns N-cols [16w,16w+16), K=128 -> 4 WMMA.
// ---------------------------------------------------------------------------
__global__ __launch_bounds__(256) void rnn_pre_kernel(
    const float* __restrict__ emb, const float* __restrict__ W_b,
    const __bf16* __restrict__ wbot, float* __restrict__ preR) {
  int lane = threadIdx.x & 31, w = threadIdx.x >> 5;
  int r = blockIdx.x >> 5;
  int bbase = (blockIdx.x & 31) * 16;
  int arow = lane & 15, khalf = (lane >> 4) << 3;
  int col = 16 * w + (lane & 15), chalf = (lane >> 4) << 3;

  const v16bf* wb = (const v16bf*)wbot;
  v16bf Bt[4];
#pragma unroll
  for (int t = 0; t < 4; ++t) Bt[t] = wb[(w * 4 + t) * 32 + lane];

  const float* rowp = emb + ((size_t)(bbase + arow) * LL + r) * EE;
  v8f c;
  float bias = W_b[col];
#pragma unroll
  for (int v = 0; v < 8; ++v) c[v] = bias;
#pragma unroll
  for (int t = 0; t < 4; ++t) c = WMMA_BF16(packA(rowp, 32 * t + khalf), Bt[t], c);

  float* outp = preR + ((size_t)r * BB + bbase) * EE + col;
#pragma unroll
  for (int v = 0; v < 8; ++v) outp[(size_t)(v + chalf) * EE] = c[v];
}

// ---------------------------------------------------------------------------
// Kernel 2: sequential chain. 32 blocks x 256 threads; block owns 16 batch rows.
// node_j = bf16(node_{j-1}) @ W_top + preR[j+1]; wave 0 also projects through G.
// ---------------------------------------------------------------------------
__global__ __launch_bounds__(256) void rnn_recur_kernel(
    const float* __restrict__ emb, const float* __restrict__ G_b,
    const int* __restrict__ left_idx, const int* __restrict__ right_idx,
    const __bf16* __restrict__ wtop, const __bf16* __restrict__ gpad,
    const float* __restrict__ preR, float* __restrict__ out) {
  __shared__ alignas(16) __bf16 nodebuf[2][16 * EE];

  int lane = threadIdx.x & 31, w = threadIdx.x >> 5;
  int bbase = blockIdx.x * 16;
  int arow = lane & 15, khalf = (lane >> 4) << 3;
  int n = lane & 15, col = 16 * w + n, chalf = (lane >> 4) << 3;

  const v16bf* wt = (const v16bf*)wtop;
  v16bf Bt[4];
#pragma unroll
  for (int t = 0; t < 4; ++t) Bt[t] = wt[(w * 4 + t) * 32 + lane];

  v16bf Gt[4];
  float gb = 0.0f;
  if (w == 0) {
#pragma unroll
    for (int t = 0; t < 4; ++t) Gt[t] = ((const v16bf*)gpad)[t * 32 + lane];
    if (n < OPN) gb = G_b[n];
  }

  const int l0 = left_idx[0];  // chain tree: only step 0's left child is a leaf

  for (int j = 0; j < II; ++j) {
    const int r = right_idx[j];
    // C init = preR slice (W_bot term + bias, precomputed)
    const float* pr = preR + ((size_t)r * BB + bbase) * EE + col;
    v8f c;
#pragma unroll
    for (int v = 0; v < 8; ++v) c[v] = pr[(size_t)(v + chalf) * EE];

    if (j + 1 < II) {  // prefetch next step's preR slab
      int rn = right_idx[j + 1];
      __builtin_prefetch(preR + ((size_t)rn * BB + bbase) * EE + col, 0, 0);
    }

    // A operand: leaf embedding at step 0, previous node (LDS, bf16) after.
    v16bf A0, A1, A2, A3;
    if (j == 0) {
      const float* rowp = emb + ((size_t)(bbase + arow) * LL + l0) * EE;
      A0 = packA(rowp, 0 + khalf);  A1 = packA(rowp, 32 + khalf);
      A2 = packA(rowp, 64 + khalf); A3 = packA(rowp, 96 + khalf);
    } else {
      const __bf16* nb = nodebuf[(j - 1) & 1];
      A0 = ldsA(nb, arow, 0 + khalf);  A1 = ldsA(nb, arow, 32 + khalf);
      A2 = ldsA(nb, arow, 64 + khalf); A3 = ldsA(nb, arow, 96 + khalf);
    }

    c = WMMA_BF16(A0, Bt[0], c);
    c = WMMA_BF16(A1, Bt[1], c);
    c = WMMA_BF16(A2, Bt[2], c);
    c = WMMA_BF16(A3, Bt[3], c);

    // publish node_j as bf16 (next step's A operand + projection input)
    __bf16* sb = nodebuf[j & 1];
#pragma unroll
    for (int v = 0; v < 8; ++v) sb[(v + chalf) * EE + col] = (__bf16)c[v];
    __syncthreads();

    if (w == 0) {  // logits: node_j @ Gpad (+ G_b via C init), 5 valid columns
      const __bf16* nb = nodebuf[j & 1];
      v8f oc;
#pragma unroll
      for (int v = 0; v < 8; ++v) oc[v] = gb;
      oc = WMMA_BF16(ldsA(nb, arow, 0 + khalf),  Gt[0], oc);
      oc = WMMA_BF16(ldsA(nb, arow, 32 + khalf), Gt[1], oc);
      oc = WMMA_BF16(ldsA(nb, arow, 64 + khalf), Gt[2], oc);
      oc = WMMA_BF16(ldsA(nb, arow, 96 + khalf), Gt[3], oc);
      if (n < OPN) {
#pragma unroll
        for (int v = 0; v < 8; ++v) {
          int b = bbase + v + chalf;
          out[((size_t)b * II + j) * OPN + n] = oc[v];
        }
      }
    }
  }
}

extern "C" void kernel_launch(void* const* d_in, const int* in_sizes, int n_in,
                              void* d_out, int out_size, void* d_ws, size_t ws_size,
                              hipStream_t stream) {
  const float* emb   = (const float*)d_in[0];  // [B, L, E]
  const float* W_w   = (const float*)d_in[1];  // [2E, E]
  const float* W_b   = (const float*)d_in[2];  // [E]
  const float* G_w   = (const float*)d_in[3];  // [E, OP]
  const float* G_b   = (const float*)d_in[4];  // [OP]
  const int*   li    = (const int*)d_in[5];    // [I]
  const int*   ri    = (const int*)d_in[6];    // [I]

  // workspace layout
  float*  preR = (float*)d_ws;                                   // L*B*E f32
  size_t  preB = (size_t)LL * BB * EE * sizeof(float);
  __bf16* wtop = (__bf16*)((char*)d_ws + preB);                  // 16384 bf16
  __bf16* wbot = wtop + 8 * 4 * 32 * 16;                         // 16384 bf16
  __bf16* gpad = wbot + 8 * 4 * 32 * 16;                         // 2048 bf16

  rnn_prep_kernel<<<64, 256, 0, stream>>>(W_w, G_w, wtop, wbot, gpad);
  rnn_pre_kernel<<<LL * (BB / 16), 256, 0, stream>>>(emb, W_b, wbot, preR);
  rnn_recur_kernel<<<BB / 16, 256, 0, stream>>>(emb, G_b, li, ri, wtop, gpad,
                                                preR, (float*)d_out);

  // second tuple output: labels passthrough (raw bytes after node_prob)
  size_t np = (size_t)BB * II * OPN;
  if ((size_t)out_size > np) {
    size_t rem = ((size_t)out_size - np) * sizeof(float);
    size_t lab = (size_t)II * sizeof(long long);
    hipMemcpyAsync((char*)d_out + np * sizeof(float), d_in[7],
                   rem < lab ? rem : lab, hipMemcpyDeviceToDevice, stream);
  }
}
